// ImagePoolingAttentionModule_24653112279095
// MI455X (gfx1250) — matrix operand order repaired and checked by reading, not compiled, not run on hardware
//
#include <hip/hip_runtime.h>
#include <hip/hip_bf16.h>

// ---------------------------------------------------------------------------
// ImagePoolingAttention for MI455X (gfx1250, wave32, WMMA,
// double-buffered async global->LDS staging, transposed bf16 GEMM weights).
// ---------------------------------------------------------------------------

typedef __attribute__((ext_vector_type(16))) __bf16 bf16x16;
typedef __attribute__((ext_vector_type(8)))  __bf16 bf16x8;
typedef __attribute__((ext_vector_type(8)))  float  v8f;

#define E_DIM   256
#define T_DIM   512
#define NTOK    80
#define KVTOK   27
#define BATCH   8
#define NEG_INF (-3.402823466e38f)

#define KB      256                 // K-block staged per pipeline stage (fp32)
#define KPAD    4                   // LDS row padding (floats) to spread banks
#define KROW    (KB + KPAD)         // LDS row stride
#define TILE    (16 * KROW)         // floats per LDS buffer
#define CNT     16                  // async b32 ops per thread per stage (KB*16/256)

// ---- gfx1250 async global->LDS copy (guarded; falls back to plain store) ---
#if defined(__has_builtin)
#if __has_builtin(__builtin_amdgcn_global_load_async_to_lds_b32)
#define IPA_HAS_ASYNC_LDS 1
#endif
#endif

typedef __attribute__((address_space(1))) int ipa_gint;   // global
typedef __attribute__((address_space(3))) int ipa_lint;   // LDS

__device__ __forceinline__ void ipa_stage_f32(float* lds_dst, const float* gsrc)
{
#ifdef IPA_HAS_ASYNC_LDS
    __builtin_amdgcn_global_load_async_to_lds_b32(
        (ipa_gint*)gsrc, (ipa_lint*)lds_dst, 0, 0);
#else
    *lds_dst = *gsrc;
#endif
}

template<int N>
__device__ __forceinline__ void ipa_async_wait()
{
#ifdef IPA_HAS_ASYNC_LDS
#if __has_builtin(__builtin_amdgcn_s_wait_asynccnt)
    __builtin_amdgcn_s_wait_asynccnt(N);
#else
    asm volatile("s_wait_asynccnt %0" :: "i"(N) : "memory");
#endif
#endif
}

__device__ __forceinline__ void ipa_atomic_max_f32(float* addr, float val)
{
#if defined(__has_builtin) && __has_builtin(__hip_atomic_fetch_max)
    __hip_atomic_fetch_max(addr, val, __ATOMIC_RELAXED, __HIP_MEMORY_SCOPE_AGENT);
#else
    int old = __float_as_int(*addr);
    while (val > __int_as_float(old)) {
        int prev = atomicCAS((int*)addr, old, __float_as_int(val));
        if (prev == old) break;
        old = prev;
    }
#endif
}

// WMMA 16x16x32 bf16 operand index maps (lane = 0..31); A/B use this code's contiguous k order:
//   A (16xK tile): row m = lane&15 ; elem i -> k = (lane>>4)*16 + i
//   B (Kx16 tile): col n = lane&15 ; elem i -> k = (lane>>4)*16 + i
//   C/D:           VGPR r -> row m = r + 8*(lane>>4) ; col n = lane&15

// ---------------------------------------------------------------------------
// Kernel 0a: elementwise fp32 -> bf16 (row-major, for conv A-operand weights).
// ---------------------------------------------------------------------------
__global__ void ipa_cvt_bf16_kernel(const float* __restrict__ in,
                                    __bf16* __restrict__ out, int n)
{
    const int i = blockIdx.x * 256 + threadIdx.x;
    if (i < n) out[i] = (__bf16)in[i];
}

// Kernel 0b: fp32 (K,N) -> bf16 (N,K) transpose-convert (GEMM B-operand
// weights: columns become contiguous -> vector loads in the GEMM kernel).
__global__ void ipa_cvt_bf16_t_kernel(const float* __restrict__ in,
                                      __bf16* __restrict__ out, int K, int N)
{
    const int i = blockIdx.x * 256 + threadIdx.x;
    if (i < K * N) {
        const int k = i / N, n = i % N;
        out[(size_t)n * K + k] = (__bf16)in[i];
    }
}

// Kernel 0c: fill float buffer (kv init to -FLT_MAX for atomic max).
__global__ void ipa_fill_kernel(float* __restrict__ p, float val, int n)
{
    const int i = blockIdx.x * 256 + threadIdx.x;
    if (i < n) p[i] = val;
}

// ---------------------------------------------------------------------------
// Kernel 1: fused 1x1-conv + adaptive max pool (3x3 regions), pipelined.
// grid: x = 9*nsplit (region, chunk-split), y = batch, z = 2 (channel halves);
// block = 256 (8 waves). Stages of 256 input channels x 16 pooled positions
// are async-copied into double-buffered LDS; stage s+1 is in flight while all
// 8 waves run the 8 WMMAs of stage s. Partial region maxima are combined with
// device-scope float atomic-max (order-independent -> deterministic).
// ---------------------------------------------------------------------------
template<int CIN>
__global__ void ipa_conv_pool_kernel(const float* __restrict__ X,     // (B, CIN, H, W)
                                     const __bf16* __restrict__ Wb,   // (256, CIN) bf16
                                     const float* __restrict__ bias,  // (256)
                                     float* __restrict__ kv,          // (B,27,256), pre-filled -inf
                                     int H, int Wd, int t_off, int nsplit)
{
    constexpr int NKB = CIN / KB;            // K-blocks per position chunk
    extern __shared__ float lds[];           // [2][16][KROW]

    const int r     = blockIdx.x / nsplit;   // pooled region 0..8
    const int split = blockIdx.x % nsplit;
    const int b     = blockIdx.y;
    const int ri  = r / 3, rj = r % 3;
    const int hs  = (ri * H) / 3;
    const int he  = ((ri + 1) * H + 2) / 3;
    const int ws0 = (rj * Wd) / 3;
    const int we  = ((rj + 1) * Wd + 2) / 3;
    const int wcnt = we - ws0;
    const int area = (he - hs) * wcnt;

    const int tid  = threadIdx.x;
    const int lane = tid & 31;
    const int wave = tid >> 5;
    const int o0   = blockIdx.z * 128 + wave * 16;  // this wave's 16-channel tile
    const int m    = lane & 15;
    const int g    = lane >> 4;

    // staging role: fixed position & channel base per thread (256 = 16*16)
    const int pos   = tid & 15;
    const int cbase = tid >> 4;              // 0..15

    v8f cinit;
#pragma unroll
    for (int rr = 0; rr < 8; ++rr) cinit[rr] = bias[o0 + rr + 8 * g];

    v8f maxacc;
#pragma unroll
    for (int rr = 0; rr < 8; ++rr) maxacc[rr] = NEG_INF;

    const size_t cstride = (size_t)H * Wd;
    const float* Xb = X + (size_t)b * CIN * cstride;

    const int CH   = (area + 15) >> 4;                                // total 16-pos chunks
    const int mych = (CH > split) ? (CH - split + nsplit - 1) / nsplit : 0;
    const int S    = mych * NKB;                                      // pipeline stages

    // issue all CNT async copies for stage s into buffer dbuf
    auto issue_stage = [&](int s, float* dbuf) {
        const int j  = s / NKB;
        const int kb = (s % NKB) * KB;
        const int p0 = (split + j * nsplit) * 16;
        int p = p0 + pos;
        if (p > area - 1) p = area - 1;      // tail clamp (dup position is max-safe)
        const int off = (hs + p / wcnt) * Wd + ws0 + p % wcnt;
        float* drow = dbuf + pos * KROW + cbase;
        const float* srow = Xb + off + (size_t)(kb + cbase) * cstride;
#pragma unroll
        for (int jj = 0; jj < CNT; ++jj)
            ipa_stage_f32(drow + jj * 16, srow + (size_t)(jj * 16) * cstride);
    };

    if (S > 0) issue_stage(0, lds);

    v8f acc = cinit;
    for (int s = 0; s < S; ++s) {
        if (s % NKB == 0) acc = cinit;

        if (s + 1 < S) {
            issue_stage(s + 1, lds + ((s + 1) & 1) * TILE);
            ipa_async_wait<CNT>();           // stage s landed; s+1 still in flight
        } else {
            ipa_async_wait<0>();
        }
        __syncthreads();

        const float*  bt = lds + (s & 1) * TILE + m * KROW;
        const __bf16* wr = Wb + (size_t)(o0 + m) * CIN + (s % NKB) * KB;
#pragma unroll
        for (int c0 = 0; c0 < KB; c0 += 32) {
            bf16x16 a;
            const bf16x8 alo = *(const bf16x8*)(wr + c0 + g * 16);
            const bf16x8 ahi = *(const bf16x8*)(wr + c0 + g * 16 + 8);
#pragma unroll
            for (int i = 0; i < 8; ++i) { a[i] = alo[i]; a[8 + i] = ahi[i]; }

            bf16x16 bm;
            const float* brow = bt + c0 + g * 16;
#pragma unroll
            for (int i = 0; i < 16; ++i) bm[i] = (__bf16)brow[i];

            acc = __builtin_amdgcn_wmma_f32_16x16x32_bf16(
                false, a, false, bm, (short)0, acc, false, false);
        }
        __syncthreads();                     // buffer reusable two stages later

        if (s % NKB == NKB - 1) {
#pragma unroll
            for (int rr = 0; rr < 8; ++rr) maxacc[rr] = fmaxf(maxacc[rr], acc[rr]);
        }
    }

    // Max over the 16 positions held per 16-lane group (same output channel).
#pragma unroll
    for (int rr = 0; rr < 8; ++rr) {
        float v = maxacc[rr];
        v = fmaxf(v, __shfl_xor(v, 1, 32));
        v = fmaxf(v, __shfl_xor(v, 2, 32));
        v = fmaxf(v, __shfl_xor(v, 4, 32));
        v = fmaxf(v, __shfl_xor(v, 8, 32));
        maxacc[rr] = v;
    }
    if ((lane & 15) == 0) {
        const int t = t_off + r;
        float* dst = kv + ((size_t)b * KVTOK + t) * E_DIM + o0 + 8 * g;
#pragma unroll
        for (int rr = 0; rr < 8; ++rr) ipa_atomic_max_f32(&dst[rr], maxacc[rr]);
    }
}

// ---------------------------------------------------------------------------
// Kernel 2: LayerNorm over T=512 for text -> bf16. grid=640 rows, block=256.
// ---------------------------------------------------------------------------
__global__ void ipa_ln_text_kernel(const float* __restrict__ x,
                                   const float* __restrict__ g,
                                   const float* __restrict__ b,
                                   __bf16* __restrict__ out)
{
    __shared__ float red[256];
    const int row = blockIdx.x, tid = threadIdx.x;
    const float* xr = x + (size_t)row * T_DIM;
    const float v0 = xr[tid], v1 = xr[tid + 256];
    red[tid] = v0 + v1;
    __syncthreads();
    for (int o = 128; o > 0; o >>= 1) { if (tid < o) red[tid] += red[tid + o]; __syncthreads(); }
    const float mu = red[0] * (1.0f / T_DIM);
    __syncthreads();
    const float c0 = v0 - mu, c1 = v1 - mu;
    red[tid] = c0 * c0 + c1 * c1;
    __syncthreads();
    for (int o = 128; o > 0; o >>= 1) { if (tid < o) red[tid] += red[tid + o]; __syncthreads(); }
    const float rstd = rsqrtf(red[0] * (1.0f / T_DIM) + 1e-5f);
    out[(size_t)row * T_DIM + tid]       = (__bf16)(c0 * rstd * g[tid] + b[tid]);
    out[(size_t)row * T_DIM + tid + 256] = (__bf16)(c1 * rstd * g[tid + 256] + b[tid + 256]);
}

// ---------------------------------------------------------------------------
// Kernel 3: LayerNorm over E=256 for kv; shared stats, two affine outputs.
// ---------------------------------------------------------------------------
__global__ void ipa_ln_kv_kernel(const float* __restrict__ kv,
                                 const float* __restrict__ kg, const float* __restrict__ kb,
                                 const float* __restrict__ vg, const float* __restrict__ vb,
                                 __bf16* __restrict__ outk, __bf16* __restrict__ outv)
{
    __shared__ float red[256];
    const int row = blockIdx.x, tid = threadIdx.x;
    const float v = kv[(size_t)row * E_DIM + tid];
    red[tid] = v;
    __syncthreads();
    for (int o = 128; o > 0; o >>= 1) { if (tid < o) red[tid] += red[tid + o]; __syncthreads(); }
    const float mu = red[0] * (1.0f / E_DIM);
    __syncthreads();
    const float c = v - mu;
    red[tid] = c * c;
    __syncthreads();
    for (int o = 128; o > 0; o >>= 1) { if (tid < o) red[tid] += red[tid + o]; __syncthreads(); }
    const float rstd = rsqrtf(red[0] * (1.0f / E_DIM) + 1e-5f);
    const float nrm = c * rstd;
    outk[(size_t)row * E_DIM + tid] = (__bf16)(nrm * kg[tid] + kb[tid]);
    outv[(size_t)row * E_DIM + tid] = (__bf16)(nrm * vg[tid] + vb[tid]);
}

// ---------------------------------------------------------------------------
// Kernel 4: out(M,N) = A_bf16(M,K) @ W(K,N) + bias [+ residual]
// W supplied TRANSPOSED as Wt(N,K) bf16 -> both operands are contiguous
// 16-byte vector loads. grid = (ceil(M/16), N/16), block = 32.
// ---------------------------------------------------------------------------
__global__ void ipa_wmma_gemm_kernel(const __bf16* __restrict__ A,
                                     const __bf16* __restrict__ Wt,  // (N, K) bf16
                                     const float* __restrict__ bias,
                                     const float* __restrict__ residual,
                                     float* __restrict__ out,
                                     int M, int N, int K)
{
    const int lane = threadIdx.x & 31;
    const int m = lane & 15;
    const int g = lane >> 4;
    const int m0 = blockIdx.x * 16;
    const int n0 = blockIdx.y * 16;

    v8f acc;
    const float bv = bias[n0 + m];        // C[mm][n] = bias[n], n = lane&15
#pragma unroll
    for (int rr = 0; rr < 8; ++rr) acc[rr] = bv;

    int arow = m0 + m;
    if (arow > M - 1) arow = M - 1;       // tail clamp (duplicate row; store-guarded)
    const __bf16* Ar = A  + (size_t)arow * K;
    const __bf16* Br = Wt + (size_t)(n0 + m) * K;   // row n of W^T

    for (int k0 = 0; k0 < K; k0 += 32) {
        bf16x16 a;
        const bf16x8 alo = *(const bf16x8*)(Ar + k0 + g * 16);
        const bf16x8 ahi = *(const bf16x8*)(Ar + k0 + g * 16 + 8);
#pragma unroll
        for (int i = 0; i < 8; ++i) { a[i] = alo[i]; a[8 + i] = ahi[i]; }

        bf16x16 bb;                        // elem i -> k = g*16 + i (contiguous)
        const bf16x8 blo = *(const bf16x8*)(Br + k0 + g * 16);
        const bf16x8 bhi = *(const bf16x8*)(Br + k0 + g * 16 + 8);
#pragma unroll
        for (int i = 0; i < 8; ++i) { bb[i] = blo[i]; bb[8 + i] = bhi[i]; }

        acc = __builtin_amdgcn_wmma_f32_16x16x32_bf16(
            false, a, false, bb, (short)0, acc, false, false);
    }

#pragma unroll
    for (int rr = 0; rr < 8; ++rr) {
        const int rowOut = m0 + rr + 8 * g;
        if (rowOut < M) {
            const size_t idx = (size_t)rowOut * N + n0 + m;
            float v = acc[rr];
            if (residual) v += residual[idx];
            out[idx] = v;
        }
    }
}

// ---------------------------------------------------------------------------
// Kernel 5: 8-head attention (80 queries x 27 keys, head_c = 32), VALU.
// grid = B*HEADS = 64, block = 128. k/v head slices staged in LDS.
// ---------------------------------------------------------------------------
__global__ void ipa_attn_kernel(const float* __restrict__ q,   // (640, 256)
                                const float* __restrict__ k,   // (216, 256)
                                const float* __restrict__ v,   // (216, 256)
                                __bf16* __restrict__ xout)     // (640, 256)
{
    __shared__ float ks[KVTOK * 32];
    __shared__ float vs[KVTOK * 32];
    const int bh = blockIdx.x;
    const int b = bh >> 3, h = bh & 7;
    const int tid = threadIdx.x;

    for (int i = tid; i < KVTOK * 32; i += blockDim.x) {
        const int t = i >> 5, c = i & 31;
        const size_t src = ((size_t)b * KVTOK + t) * E_DIM + h * 32 + c;
        ks[i] = k[src];
        vs[i] = v[src];
    }
    __syncthreads();

    if (tid < NTOK) {
        float qr[32];
        const float* qrow = q + ((size_t)b * NTOK + tid) * E_DIM + h * 32;
#pragma unroll
        for (int c = 0; c < 32; ++c) qr[c] = qrow[c];

        float s[KVTOK];
        float mx = NEG_INF;
        for (int t = 0; t < KVTOK; ++t) {
            float d = 0.0f;
#pragma unroll
            for (int c = 0; c < 32; ++c) d += qr[c] * ks[t * 32 + c];
            d *= 0.17677669529663689f;   // 1/sqrt(32)
            s[t] = d;
            mx = fmaxf(mx, d);
        }
        float sum = 0.0f;
        for (int t = 0; t < KVTOK; ++t) { s[t] = __expf(s[t] - mx); sum += s[t]; }
        const float inv = 1.0f / sum;

        __bf16* xo = xout + ((size_t)b * NTOK + tid) * E_DIM + h * 32;
#pragma unroll
        for (int c = 0; c < 32; ++c) {
            float o = 0.0f;
            for (int t = 0; t < KVTOK; ++t) o += s[t] * vs[t * 32 + c];
            xo[c] = (__bf16)(o * inv);
        }
    }
}

// ---------------------------------------------------------------------------
// Host-side launch
// ---------------------------------------------------------------------------
extern "C" void kernel_launch(void* const* d_in, const int* in_sizes, int n_in,
                              void* d_out, int out_size, void* d_ws, size_t ws_size,
                              hipStream_t stream)
{
    (void)in_sizes; (void)n_in; (void)out_size; (void)ws_size;

    const float* text = (const float*)d_in[0];
    const float* img0 = (const float*)d_in[1];
    const float* img1 = (const float*)d_in[2];
    const float* img2 = (const float*)d_in[3];
    const float* cw0  = (const float*)d_in[4];
    const float* cw1  = (const float*)d_in[5];
    const float* cw2  = (const float*)d_in[6];
    const float* cb0  = (const float*)d_in[7];
    const float* cb1  = (const float*)d_in[8];
    const float* cb2  = (const float*)d_in[9];
    const float* qlng = (const float*)d_in[10];
    const float* qlnb = (const float*)d_in[11];
    const float* klng = (const float*)d_in[12];
    const float* klnb = (const float*)d_in[13];
    const float* vlng = (const float*)d_in[14];
    const float* vlnb = (const float*)d_in[15];
    const float* qw   = (const float*)d_in[16];
    const float* qb   = (const float*)d_in[17];
    const float* kw   = (const float*)d_in[18];
    const float* kb   = (const float*)d_in[19];
    const float* vw   = (const float*)d_in[20];
    const float* vb   = (const float*)d_in[21];
    const float* pw   = (const float*)d_in[22];
    const float* pb   = (const float*)d_in[23];
    float* out = (float*)d_out;

    // Workspace carve-up (offsets 256B-aligned).
    char* ws = (char*)d_ws;
    float*  kvbuf  = (float*) (ws + 0);        // 8*27*256 f32 = 221184 B
    __bf16* textln = (__bf16*)(ws + 221184);   // 640*512     = 655360 B
    __bf16* kvln_k = (__bf16*)(ws + 876544);   // 216*256     = 110592 B
    __bf16* kvln_v = (__bf16*)(ws + 987136);   // 216*256     = 110592 B
    float*  qbuf   = (float*) (ws + 1097728);  // 640*256 f32 = 655360 B
    float*  kbuf   = (float*) (ws + 1753088);  // 216*256 f32 = 221184 B
    float*  vbuf   = (float*) (ws + 1974272);  // 216*256 f32 = 221184 B
    __bf16* xbf    = (__bf16*)(ws + 2195456);  // 640*256     = 327680 B
    __bf16* cw0b   = (__bf16*)(ws + 2523136);  // 65536       = 131072 B
    __bf16* cw1b   = (__bf16*)(ws + 2654208);  // 131072      = 262144 B
    __bf16* cw2b   = (__bf16*)(ws + 2916352);  // 262144      = 524288 B
    __bf16* qwt    = (__bf16*)(ws + 3440640);  // (256,512)   = 262144 B
    __bf16* kwt    = (__bf16*)(ws + 3702784);  // (256,256)   = 131072 B
    __bf16* vwt    = (__bf16*)(ws + 3833856);  // (256,256)   = 131072 B
    __bf16* pwt    = (__bf16*)(ws + 3964928);  // (512,256)   = 262144 B
    // total ~4.23 MB

    // 0) weight prep: conv weights row-major bf16 (A-operand);
    //    GEMM weights transpose-converted to (N,K) bf16 (B-operand);
    //    kv init to -FLT_MAX for atomic max.
    ipa_cvt_bf16_kernel  <<<(65536  + 255) / 256, 256, 0, stream>>>(cw0, cw0b, 65536);
    ipa_cvt_bf16_kernel  <<<(131072 + 255) / 256, 256, 0, stream>>>(cw1, cw1b, 131072);
    ipa_cvt_bf16_kernel  <<<(262144 + 255) / 256, 256, 0, stream>>>(cw2, cw2b, 262144);
    ipa_cvt_bf16_t_kernel<<<(131072 + 255) / 256, 256, 0, stream>>>(qw, qwt, 512, 256);
    ipa_cvt_bf16_t_kernel<<<(65536  + 255) / 256, 256, 0, stream>>>(kw, kwt, 256, 256);
    ipa_cvt_bf16_t_kernel<<<(65536  + 255) / 256, 256, 0, stream>>>(vw, vwt, 256, 256);
    ipa_cvt_bf16_t_kernel<<<(131072 + 255) / 256, 256, 0, stream>>>(pw, pwt, 256, 512);
    ipa_fill_kernel<<<(55296 + 255) / 256, 256, 0, stream>>>(kvbuf, NEG_INF, 55296);

    // 1) fused conv + adaptive max pool (3 scales) -> kv (8,27,256)
    const size_t smem = (size_t)2 * TILE * sizeof(float);    // 33280 B (double buffer)
    ipa_conv_pool_kernel<256 ><<<dim3(9 * 4, BATCH, 2), 256, smem, stream>>>(
        img0, cw0b, cb0, kvbuf, 80, 80, 0, 4);
    ipa_conv_pool_kernel<512 ><<<dim3(9 * 2, BATCH, 2), 256, smem, stream>>>(
        img1, cw1b, cb1, kvbuf, 40, 40, 9, 2);
    ipa_conv_pool_kernel<1024><<<dim3(9 * 1, BATCH, 2), 256, smem, stream>>>(
        img2, cw2b, cb2, kvbuf, 20, 20, 18, 1);

    // 2) LayerNorms
    ipa_ln_text_kernel<<<BATCH * NTOK, 256, 0, stream>>>(text, qlng, qlnb, textln);
    ipa_ln_kv_kernel<<<BATCH * KVTOK, 256, 0, stream>>>(kvbuf, klng, klnb, vlng, vlnb,
                                                        kvln_k, kvln_v);

    // 3) q / k / v projections (bf16 WMMA, f32 out)
    ipa_wmma_gemm_kernel<<<dim3(40, 16), 32, 0, stream>>>(textln, qwt, qb, nullptr, qbuf,
                                                          640, 256, 512);
    ipa_wmma_gemm_kernel<<<dim3(14, 16), 32, 0, stream>>>(kvln_k, kwt, kb, nullptr, kbuf,
                                                          216, 256, 256);
    ipa_wmma_gemm_kernel<<<dim3(14, 16), 32, 0, stream>>>(kvln_v, vwt, vb, nullptr, vbuf,
                                                          216, 256, 256);

    // 4) softmax attention (VALU; tiny)
    ipa_attn_kernel<<<BATCH * 8, 128, 0, stream>>>(qbuf, kbuf, vbuf, xbf);

    // 5) output projection + bias + residual (WMMA)
    ipa_wmma_gemm_kernel<<<dim3(40, 32), 32, 0, stream>>>(xbf, pwt, pb, text, out,
                                                          640, 512, 256);
}
